// Sg2ImModel_32332513804839
// MI455X (gfx1250) — compile-verified, matrix operand order checked by reading, not run on hardware
//
#include <hip/hip_runtime.h>

// =====================================================================
// Sg2Im forward for MI455X (gfx1250, wave32, WMMA).
//
// Input leaf order assumption: jax.tree_util-style flatten of the
// setup_inputs() dict, dict keys sorted at every level, lists/tuples in
// order:
//   0: boxes_gt (384*4 f32)        1: num_imgs (1 i32, ==24)
//   2: obj_to_img (384 i32)        3: objs (384 i32)
//   4..7:   params.box_net   [W1(128,512), b1, W2(512,4), b2]
//   8..47:  params.gconvs[g] (base 8+8g):
//             n1W1(3din,512), n1b1, n1W2(512,1152), n1b2,
//             n2W1(512,512),  n2b1, n2W2(512,128),  n2b2
//   48:     params.obj_emb (185,64)
//   49..52: params.out_conv [W1(64,64,3,3), W2(3,64,1,1), b1(64), b2(3)]
//   53:     params.pred_emb (46,64)
//   54..93: params.refine[m] (base 54+8m): W1, W2, b1, b2, be1, be2, g1, g2
//   94..97: params.rel_aux [W1(136,512), b1, W2(512,46), b2]
//   98:     triples (768*3 i32)
// Output: img(24*3*64*64) || boxes_pred(384*4) || rel_scores(768*46), f32.
// =====================================================================

#define LEAKY 0.2f
#define BNEPS 1e-5f
#define N_IMG 24
#define GHID 512
#define GDIM 128

typedef __attribute__((ext_vector_type(16))) _Float16 v16h;
typedef __attribute__((ext_vector_type(8)))  float    v8f;

static __device__ __forceinline__ float actf(float v, int act) {
  if (act == 1) return v > 0.f ? v : 0.f;          // relu
  if (act == 2) return v >= 0.f ? v : LEAKY * v;   // leaky
  return v;
}

static __device__ __forceinline__ void cvt8(v16h& d, int base, float4 p, float4 q) {
  d[base + 0] = (_Float16)p.x; d[base + 1] = (_Float16)p.y;
  d[base + 2] = (_Float16)p.z; d[base + 3] = (_Float16)p.w;
  d[base + 4] = (_Float16)q.x; d[base + 5] = (_Float16)q.y;
  d[base + 6] = (_Float16)q.z; d[base + 7] = (_Float16)q.w;
}

// LDS fragment swizzle for one 32(K)x16(N) f16 chunk: element (k,n) lives at
// lane = n + 16*k[3], slot = 8*k[4] | k[2:0]  (matches the CDNA5 16-bit B
// 32x16 VGPR layout), so each lane reads its 16-half fragment contiguously
// (2x ds_load_b128).
static __device__ __forceinline__ int bswz(int kk, int nn) {
  return (nn + ((kk >> 3) & 1) * 16) * 16 + (((kk >> 4) << 3) | (kk & 7));
}

// -------------------- WMMA GEMM: C = act(A[M,K] @ W[K,N] + b) ----------
// Requirements (guaranteed by all call sites): M % 128 == 0, K % 4 == 0,
// A/W/C 16B aligned. N may be ragged (guarded slow staging path + guarded
// stores). Block = 4 waves; each wave owns a 32x16 C tile; the block shares
// one B chunk staged in LDS per 32-wide K step. Steady-state per chunk:
// staging = 1x global_load_b128/thread (branch-free), compute = 8x
// global_load_b128 (A) + 2x ds_load_b128 (B frag) + 2x wmma per wave.
__global__ void wmma_gemm_bias_act(const float* __restrict__ A,
                                   const float* __restrict__ W,
                                   const float* __restrict__ bias,
                                   float* __restrict__ C,
                                   int M, int N, int K, int act) {
  __shared__ __align__(32) _Float16 Bsh[512];
  const int tid  = threadIdx.x;
  const int wave = tid >> 5;
  const int lane = tid & 31;
  const int half = lane >> 4;
  const int lr   = lane & 15;
  const int col0 = blockIdx.x * 16;
  const int col  = col0 + lr;
  const int m0   = (blockIdx.y * 4 + wave) * 32;
  const int rowA0 = m0 + lr;
  const int rowA1 = m0 + 16 + lr;
  const bool fastN = ((N & 3) == 0) && (col0 + 16 <= N);
  // fast staging geometry: thread -> one float4 of row kk
  const int st_kk  = tid >> 2;        // 0..31
  const int st_nn0 = (tid & 3) * 4;   // 0,4,8,12
  const int st_hib = ((st_kk >> 3) & 1) * 16;
  const int st_slot = ((st_kk >> 4) << 3) | (st_kk & 7);
  v8f acc0 = {}, acc1 = {};
  for (int k0 = 0; k0 < K; k0 += 32) {
    __syncthreads();
    if (fastN && (k0 + 32 <= K)) {
      const float4 wv = *(const float4*)&W[(long)(k0 + st_kk) * N + col0 + st_nn0];
      Bsh[(st_nn0 + 0 + st_hib) * 16 + st_slot] = (_Float16)wv.x;
      Bsh[(st_nn0 + 1 + st_hib) * 16 + st_slot] = (_Float16)wv.y;
      Bsh[(st_nn0 + 2 + st_hib) * 16 + st_slot] = (_Float16)wv.z;
      Bsh[(st_nn0 + 3 + st_hib) * 16 + st_slot] = (_Float16)wv.w;
    } else {
#pragma unroll
      for (int e = tid; e < 512; e += 128) {
        const int kk = e >> 4, nn = e & 15;
        const int k = k0 + kk, n = col0 + nn;
        const float v = (k < K && n < N) ? W[(long)k * N + n] : 0.f;
        Bsh[bswz(kk, nn)] = (_Float16)v;
      }
    }
    __syncthreads();
    const v16h bfrag = *((const v16h*)&Bsh[lane * 16]);
    v16h a0f, a1f;
    if (k0 + 32 <= K) {
      const float* ap0 = A + (long)rowA0 * K + k0 + half * 8;
      const float* ap1 = A + (long)rowA1 * K + k0 + half * 8;
      if (k0 + 32 < K) {
        __builtin_prefetch(ap0 + 32, 0, 3);  // global_prefetch_b8
        __builtin_prefetch(ap1 + 32, 0, 3);
      }
      cvt8(a0f, 0, *(const float4*)(ap0),      *(const float4*)(ap0 + 4));
      cvt8(a0f, 8, *(const float4*)(ap0 + 16), *(const float4*)(ap0 + 20));
      cvt8(a1f, 0, *(const float4*)(ap1),      *(const float4*)(ap1 + 4));
      cvt8(a1f, 8, *(const float4*)(ap1 + 16), *(const float4*)(ap1 + 20));
    } else {  // ragged K tail (e.g. K=136)
#pragma unroll
      for (int s = 0; s < 16; ++s) {
        const int k = k0 + half * 8 + ((s < 8) ? s : s + 8);
        a0f[s] = (_Float16)((k < K) ? A[(long)rowA0 * K + k] : 0.f);
        a1f[s] = (_Float16)((k < K) ? A[(long)rowA1 * K + k] : 0.f);
      }
    }
    acc0 = __builtin_amdgcn_wmma_f32_16x16x32_f16(false, a0f, false, bfrag,
                                                  (short)0, acc0, false, false);
    acc1 = __builtin_amdgcn_wmma_f32_16x16x32_f16(false, a1f, false, bfrag,
                                                  (short)0, acc1, false, false);
  }
  if (col < N) {
    const float bv = bias ? bias[col] : 0.f;
#pragma unroll
    for (int v = 0; v < 8; ++v) {
      const int r0 = m0 + v + 8 * half;
      C[(long)r0 * N + col] = actf(acc0[v] + bv, act);
      C[(long)(r0 + 16) * N + col] = actf(acc1[v] + bv, act);
    }
  }
}

// ---------------- WMMA implicit-GEMM conv (NCHW, OIHW) -----------------
// M = Nb*H*W (always % 128 == 0 here), K = Ci*ks*ks, N = Co (ragged ok).
// Weight layout is contiguous in the GEMM K index per output channel
// (((co*Ci+c)*ks+dy)*ks+dx == co*K + k), so staging needs no decode and the
// fast path is branch-free coalesced loads. Gathered input side keeps only
// the zero-pad selects (inherent to implicit conv).
__global__ void wmma_conv_bias_act(const float* __restrict__ x,
                                   const float* __restrict__ w,
                                   const float* __restrict__ bias,
                                   float* __restrict__ y,
                                   int Nb, int Ci, int H, int Wd, int Co,
                                   int ks, int pad, int act) {
  __shared__ __align__(32) _Float16 Bsh[512];
  const int HW = H * Wd;
  const int K = Ci * ks * ks;
  const int ks2 = ks * ks;
  const int tid  = threadIdx.x;
  const int wave = tid >> 5;
  const int lane = tid & 31;
  const int half = lane >> 4;
  const int lr   = lane & 15;
  const int co0 = blockIdx.x * 16;
  const int co  = co0 + lr;
  const int m0  = (blockIdx.y * 4 + wave) * 32;
  const int mA0 = m0 + lr, mA1 = m0 + 16 + lr;
  const int n0 = mA0 / HW, q0 = mA0 - n0 * HW, oy0 = q0 / Wd, ox0 = q0 - oy0 * Wd;
  const int n1 = mA1 / HW, q1 = mA1 - n1 * HW, oy1 = q1 / Wd, ox1 = q1 - oy1 * Wd;
  const bool fastCo = (co0 + 16 <= Co);
  v8f acc0 = {}, acc1 = {};
  for (int k0 = 0; k0 < K; k0 += 32) {
    __syncthreads();
    if (fastCo && (k0 + 32 <= K)) {
      // k-fastest mapping: consecutive lanes read consecutive k (coalesced)
#pragma unroll
      for (int e = tid; e < 512; e += 128) {
        const int kk = e & 31, nn = e >> 5;
        Bsh[bswz(kk, nn)] = (_Float16)w[(long)(co0 + nn) * K + k0 + kk];
      }
    } else {
#pragma unroll
      for (int e = tid; e < 512; e += 128) {
        const int kk = e & 31, nn = e >> 5;
        const int k = k0 + kk, n = co0 + nn;
        const float v = (k < K && n < Co) ? w[(long)n * K + k] : 0.f;
        Bsh[bswz(kk, nn)] = (_Float16)v;
      }
    }
    __syncthreads();
    const v16h bfrag = *((const v16h*)&Bsh[lane * 16]);
    v16h a0f, a1f;
#pragma unroll
    for (int s = 0; s < 16; ++s) {
      const int k = k0 + half * 8 + ((s < 8) ? s : s + 8);
      float v0 = 0.f, v1 = 0.f;
      if (k < K) {
        const int c = k / ks2, r2 = k - c * ks2;
        const int dy = r2 / ks, dx = r2 - dy * ks;
        const int iy0 = oy0 + dy - pad, ix0 = ox0 + dx - pad;
        if (iy0 >= 0 && iy0 < H && ix0 >= 0 && ix0 < Wd)
          v0 = x[((long)(n0 * Ci + c) * H + iy0) * Wd + ix0];
        const int iy1 = oy1 + dy - pad, ix1 = ox1 + dx - pad;
        if (iy1 >= 0 && iy1 < H && ix1 >= 0 && ix1 < Wd)
          v1 = x[((long)(n1 * Ci + c) * H + iy1) * Wd + ix1];
      }
      a0f[s] = (_Float16)v0;
      a1f[s] = (_Float16)v1;
    }
    acc0 = __builtin_amdgcn_wmma_f32_16x16x32_f16(false, a0f, false, bfrag,
                                                  (short)0, acc0, false, false);
    acc1 = __builtin_amdgcn_wmma_f32_16x16x32_f16(false, a1f, false, bfrag,
                                                  (short)0, acc1, false, false);
  }
  if (co < Co) {
    const float bb = bias ? bias[co] : 0.f;
#pragma unroll
    for (int v = 0; v < 8; ++v) {
      const int mm0 = m0 + v + 8 * half;
      const int nn0 = mm0 / HW, rr0 = mm0 - nn0 * HW;
      y[(long)(nn0 * Co + co) * HW + rr0] = actf(acc0[v] + bb, act);
      const int mm1 = mm0 + 16;
      const int nn1 = mm1 / HW, rr1 = mm1 - nn1 * HW;
      y[(long)(nn1 * Co + co) * HW + rr1] = actf(acc1[v] + bb, act);
    }
  }
}

// ------------------------- scalar glue kernels -------------------------
__global__ void k_zero(float* p, int n) {
  int i = blockIdx.x * blockDim.x + threadIdx.x;
  if (i < n) p[i] = 0.f;
}

__global__ void k_embed(const float* __restrict__ tab, const int* __restrict__ idx,
                        int istride, int ioff, float* __restrict__ out, int n, int dim) {
  int i = blockIdx.x * blockDim.x + threadIdx.x;
  if (i >= n * dim) return;
  int r = i / dim, d = i - r * dim;
  out[i] = tab[idx[r * istride + ioff] * dim + d];
}

__global__ void k_gather_tin(const float* __restrict__ objv, const float* __restrict__ predv,
                             const int* __restrict__ tri, float* __restrict__ t_in,
                             int nT, int din) {
  int i = blockIdx.x * blockDim.x + threadIdx.x;
  if (i >= nT * 3 * din) return;
  int t = i / (3 * din), j = i - t * 3 * din;
  float v;
  if (j < din)            v = objv[tri[t * 3] * din + j];
  else if (j < 2 * din)   v = predv[t * din + (j - din)];
  else                    v = objv[tri[t * 3 + 2] * din + (j - 2 * din)];
  t_in[i] = v;
}

__global__ void k_edge_pool(const float* __restrict__ newt, const int* __restrict__ tri,
                            float* pooled, float* cnt, int nT) {
  int i = blockIdx.x * blockDim.x + threadIdx.x;
  if (i >= nT * GHID) return;
  int t = i / GHID, j = i - t * GHID;
  int s = tri[t * 3], o = tri[t * 3 + 2];
  atomicAdd(&pooled[s * GHID + j], newt[t * 1152 + j]);
  atomicAdd(&pooled[o * GHID + j], newt[t * 1152 + GHID + GDIM + j]);
  if (j == 0) { atomicAdd(&cnt[s], 1.f); atomicAdd(&cnt[o], 1.f); }
}

__global__ void k_pool_div(float* pooled, const float* cnt, int nObj) {
  int i = blockIdx.x * blockDim.x + threadIdx.x;
  if (i >= nObj * GHID) return;
  pooled[i] /= fmaxf(cnt[i / GHID], 1.f);
}

__global__ void k_copy_pred(const float* __restrict__ newt, float* pred, int nT) {
  int i = blockIdx.x * blockDim.x + threadIdx.x;
  if (i >= nT * GDIM) return;
  int t = i / GDIM, j = i - t * GDIM;
  pred[i] = newt[t * 1152 + GHID + j];
}

__global__ void k_gather_rel(const float* __restrict__ bp, const float* __restrict__ orig,
                             const int* __restrict__ tri, float* rel, int nT) {
  int i = blockIdx.x * blockDim.x + threadIdx.x;
  if (i >= nT * 136) return;
  int t = i / 136, j = i - t * 136;
  int s = tri[t * 3], o = tri[t * 3 + 2];
  float v;
  if (j < 4)        v = bp[s * 4 + j];
  else if (j < 8)   v = bp[o * 4 + (j - 4)];
  else if (j < 72)  v = orig[s * 64 + (j - 8)];
  else              v = orig[o * 64 + (j - 72)];
  rel[i] = v;
}

static __device__ __forceinline__ float w1d(float g) {
  float p = (g + 1.f) * 0.5f * 7.f;     // IN_SIZE-1 == 7
  float i0 = floorf(p);
  float f = p - i0;
  float in0 = (i0 >= 0.f && i0 <= 7.f) ? 1.f : 0.f;
  float in1 = (i0 + 1.f >= 0.f && i0 + 1.f <= 7.f) ? 1.f : 0.f;
  return (1.f - f) * in0 + f * in1;
}

__global__ void k_layout(const float* __restrict__ vecs, const float* __restrict__ boxes,
                         const int* __restrict__ o2i, float* layout, int nObj) {
  int obj = blockIdx.x;
  if (obj >= nObj) return;
  int img = o2i[obj];
  float x0 = boxes[obj * 4 + 0], y0 = boxes[obj * 4 + 1];
  float x1 = boxes[obj * 4 + 2], y1 = boxes[obj * 4 + 3];
  float ww = x1 - x0, hh = y1 - y0;
  for (int p = threadIdx.x; p < 64 * 64; p += blockDim.x) {
    int yy = p >> 6, xx = p & 63;
    float gx = ((float)xx / 63.f - x0) / ww * 2.f - 1.f;
    float gy = ((float)yy / 63.f - y0) / hh * 2.f - 1.f;
    float wgt = w1d(gy) * w1d(gx);
    if (wgt != 0.f) {
      for (int c = 0; c < GDIM; ++c)
        atomicAdd(&layout[((img * GDIM + c) * 64 + yy) * 64 + xx],
                  vecs[obj * GDIM + c] * wgt);
    }
  }
}

__global__ void k_upsample2(const float* __restrict__ x, float* y, int NC, int H, int W) {
  int OH = 2 * H, OW = 2 * W;
  int i = blockIdx.x * blockDim.x + threadIdx.x;
  if (i >= NC * OH * OW) return;
  int nc = i / (OH * OW), r = i - nc * OH * OW, oy = r / OW, ox = r - oy * OW;
  y[i] = x[(nc * H + (oy >> 1)) * W + (ox >> 1)];
}

__global__ void k_avgpool(const float* __restrict__ x, float* y, int NC, int H, int W, int f) {
  int OH = H / f, OW = W / f;
  int i = blockIdx.x * blockDim.x + threadIdx.x;
  if (i >= NC * OH * OW) return;
  int nc = i / (OH * OW), r = i - nc * OH * OW, oy = r / OW, ox = r - oy * OW;
  float s = 0.f;
  for (int dy = 0; dy < f; ++dy)
    for (int dx = 0; dx < f; ++dx)
      s += x[(nc * H + oy * f + dy) * W + ox * f + dx];
  y[i] = s / (float)(f * f);
}

__global__ void k_concat2(const float* __restrict__ a, const float* __restrict__ b,
                          float* o, int Nb, int Ca, int Cb, int HW) {
  int C = Ca + Cb;
  int i = blockIdx.x * blockDim.x + threadIdx.x;
  if (i >= Nb * C * HW) return;
  int n = i / (C * HW), r = i - n * C * HW, c = r / HW, p = r - c * HW;
  o[i] = (c < Ca) ? a[(n * Ca + c) * HW + p] : b[(n * Cb + (c - Ca)) * HW + p];
}

__global__ void bn_stats(const float* __restrict__ x, float* mean, float* var,
                         int Nb, int C, int HW) {
  int c = blockIdx.x;
  __shared__ float ss[256], sq[256];
  float s = 0.f, q = 0.f;
  int total = Nb * HW;
  for (int i = threadIdx.x; i < total; i += 256) {
    int n = i / HW, r = i - n * HW;
    float v = x[(long)(n * C + c) * HW + r];
    s += v; q += v * v;
  }
  ss[threadIdx.x] = s; sq[threadIdx.x] = q;
  __syncthreads();
  for (int st = 128; st > 0; st >>= 1) {
    if (threadIdx.x < st) {
      ss[threadIdx.x] += ss[threadIdx.x + st];
      sq[threadIdx.x] += sq[threadIdx.x + st];
    }
    __syncthreads();
  }
  if (threadIdx.x == 0) {
    float m = ss[0] / total;
    mean[c] = m;
    var[c]  = sq[0] / total - m * m;
  }
}

__global__ void bn_apply_leaky(float* x, const float* mean, const float* var,
                               const float* g, const float* be, int Nb, int C, int HW) {
  int i = blockIdx.x * blockDim.x + threadIdx.x;
  if (i >= Nb * C * HW) return;
  int c = (i / HW) % C;
  float v = (x[i] - mean[c]) * rsqrtf(var[c] + BNEPS) * g[c] + be[c];
  x[i] = v >= 0.f ? v : LEAKY * v;
}

// ----------------------------- workspace layout (float offsets) --------
enum : long {
  OFF_OBJ0   = 0,          // 384*64
  OFF_PRED0  = 24576,      // 768*64
  OFF_OBJA   = 73728,      // 384*128
  OFF_OBJB   = 122880,     // 384*128
  OFF_PREDA  = 172032,     // 768*128
  OFF_TIN    = 270336,     // 768*384
  OFF_H1     = 565248,     // 768*512
  OFF_NEWT   = 958464,     // 768*1152
  OFF_POOL   = 1843200,    // 384*512
  OFF_CNT    = 2039808,    // 384
  OFF_H2     = 2040192,    // 384*512
  OFF_RELIN  = 2236800,    // 768*136
  OFF_RELH   = 2341248,    // 768*512
  OFF_MEAN   = 2734464,    // 1024
  OFF_VAR    = 2735488,    // 1024
  OFF_LAYOUT = 2736512,    // 24*128*64*64
  OFF_LAY    = 15319424,   // 24*128*32*32 (max downsample)
  OFF_UP     = 18465152,   // 24*128*64*64 (max upsampled feats / out tmp)
  OFF_CC     = 31048064,   // 24*256*64*64 (max concat)
  OFF_X      = 56213888,   // 24*64*64*64  (max conv1 out)
  OFF_FEATS  = 62505344,   // 24*64*64*64  (max feats)
};

#define EWGRID(n) dim3(((n) + 255) / 256), dim3(256), 0, stream

extern "C" void kernel_launch(void* const* d_in, const int* in_sizes, int n_in,
                              void* d_out, int out_size, void* d_ws, size_t ws_size,
                              hipStream_t stream) {
  (void)in_sizes; (void)n_in; (void)out_size; (void)ws_size;
  const float* boxes_gt   = (const float*)d_in[0];
  const int*   obj_to_img = (const int*)d_in[2];
  const int*   objs       = (const int*)d_in[3];
  const float* boxW1 = (const float*)d_in[4];
  const float* boxb1 = (const float*)d_in[5];
  const float* boxW2 = (const float*)d_in[6];
  const float* boxb2 = (const float*)d_in[7];
  const float* obj_emb  = (const float*)d_in[48];
  const float* ocW1 = (const float*)d_in[49];
  const float* ocW2 = (const float*)d_in[50];
  const float* ocb1 = (const float*)d_in[51];
  const float* ocb2 = (const float*)d_in[52];
  const float* pred_emb = (const float*)d_in[53];
  const float* relW1 = (const float*)d_in[94];
  const float* relb1 = (const float*)d_in[95];
  const float* relW2 = (const float*)d_in[96];
  const float* relb2 = (const float*)d_in[97];
  const int*   triples = (const int*)d_in[98];

  float* ws = (float*)d_ws;
  float* outp = (float*)d_out;
  const int IMG_SZ = 24 * 3 * 64 * 64;          // 294912
  float* out_boxes = outp + IMG_SZ;             // 384*4
  float* out_rel   = outp + IMG_SZ + 384 * 4;   // 768*46

  // All gemm/conv M values are multiples of 128 (384, 768, 24*H*H).
  auto gemm = [&](const float* A, const float* W, const float* b, float* C,
                  int M, int N, int K, int act) {
    wmma_gemm_bias_act<<<dim3((N + 15) / 16, M / 128), dim3(128), 0, stream>>>(
        A, W, b, C, M, N, K, act);
  };
  auto conv = [&](const float* x, const float* w, const float* b, float* y,
                  int Ci, int H, int Co, int ks, int pad, int act) {
    int M = N_IMG * H * H;
    wmma_conv_bias_act<<<dim3((Co + 15) / 16, M / 128), dim3(128), 0, stream>>>(
        x, w, b, y, N_IMG, Ci, H, H, Co, ks, pad, act);
  };
  auto bnleaky = [&](float* x, const float* g, const float* be, int C, int HW) {
    bn_stats<<<dim3(C), dim3(256), 0, stream>>>(x, ws + OFF_MEAN, ws + OFF_VAR,
                                                N_IMG, C, HW);
    int n = N_IMG * C * HW;
    bn_apply_leaky<<<EWGRID(n)>>>(x, ws + OFF_MEAN, ws + OFF_VAR, g, be,
                                  N_IMG, C, HW);
  };

  // ---- embeddings ----
  k_embed<<<EWGRID(384 * 64)>>>(obj_emb, objs, 1, 0, ws + OFF_OBJ0, 384, 64);
  k_embed<<<EWGRID(768 * 64)>>>(pred_emb, triples, 3, 1, ws + OFF_PRED0, 768, 64);

  // ---- 5 graph-conv layers ----
  const float* cur_obj  = ws + OFF_OBJ0;
  const float* cur_pred = ws + OFF_PRED0;
  int din = 64;
  for (int g = 0; g < 5; ++g) {
    int base = 8 + g * 8;
    const float* n1W1 = (const float*)d_in[base + 0];
    const float* n1b1 = (const float*)d_in[base + 1];
    const float* n1W2 = (const float*)d_in[base + 2];
    const float* n1b2 = (const float*)d_in[base + 3];
    const float* n2W1 = (const float*)d_in[base + 4];
    const float* n2b1 = (const float*)d_in[base + 5];
    const float* n2W2 = (const float*)d_in[base + 6];
    const float* n2b2 = (const float*)d_in[base + 7];

    k_gather_tin<<<EWGRID(768 * 3 * din)>>>(cur_obj, cur_pred, triples,
                                            ws + OFF_TIN, 768, din);
    gemm(ws + OFF_TIN, n1W1, n1b1, ws + OFF_H1,   768, GHID, 3 * din, 1);
    gemm(ws + OFF_H1,  n1W2, n1b2, ws + OFF_NEWT, 768, 2 * GHID + GDIM, GHID, 1);

    k_zero<<<EWGRID(384 * GHID)>>>(ws + OFF_POOL, 384 * GHID);
    k_zero<<<EWGRID(384)>>>(ws + OFF_CNT, 384);
    k_edge_pool<<<EWGRID(768 * GHID)>>>(ws + OFF_NEWT, triples,
                                        ws + OFF_POOL, ws + OFF_CNT, 768);
    k_pool_div<<<EWGRID(384 * GHID)>>>(ws + OFF_POOL, ws + OFF_CNT, 384);
    k_copy_pred<<<EWGRID(768 * GDIM)>>>(ws + OFF_NEWT, ws + OFF_PREDA, 768);

    float* obj_next = ws + ((g & 1) ? OFF_OBJB : OFF_OBJA);
    gemm(ws + OFF_POOL, n2W1, n2b1, ws + OFF_H2, 384, GHID, GHID, 1);
    gemm(ws + OFF_H2,   n2W2, n2b2, obj_next,    384, GDIM, GHID, 1);

    cur_obj = obj_next;
    cur_pred = ws + OFF_PREDA;
    din = GDIM;
  }

  // ---- box head ----
  gemm(cur_obj, boxW1, boxb1, ws + OFF_H2, 384, GHID, GDIM, 1);
  gemm(ws + OFF_H2, boxW2, boxb2, out_boxes, 384, 4, GHID, 1);

  // ---- rel head ----
  k_gather_rel<<<EWGRID(768 * 136)>>>(out_boxes, ws + OFF_OBJ0, triples,
                                      ws + OFF_RELIN, 768);
  gemm(ws + OFF_RELIN, relW1, relb1, ws + OFF_RELH, 768, GHID, 136, 1);
  gemm(ws + OFF_RELH,  relW2, relb2, out_rel,       768, 46, GHID, 1);

  // ---- boxes -> layout (24,128,64,64) ----
  k_zero<<<EWGRID(N_IMG * GDIM * 64 * 64)>>>(ws + OFF_LAYOUT, N_IMG * GDIM * 64 * 64);
  k_layout<<<dim3(384), dim3(256), 0, stream>>>(cur_obj, boxes_gt, obj_to_img,
                                                ws + OFF_LAYOUT, 384);

  // ---- refinement CNN ----
  k_zero<<<EWGRID(N_IMG * 1 * 2 * 2)>>>(ws + OFF_FEATS, N_IMG * 1 * 2 * 2);
  float* feats = ws + OFF_FEATS;
  int cinF = 1, h = 2;
  const int couts[5] = {1024, 512, 256, 128, 64};
  for (int m = 0; m < 5; ++m) {
    int base = 54 + m * 8;
    const float* W1  = (const float*)d_in[base + 0];
    const float* W2  = (const float*)d_in[base + 1];
    const float* b1  = (const float*)d_in[base + 2];
    const float* b2  = (const float*)d_in[base + 3];
    const float* be1 = (const float*)d_in[base + 4];
    const float* be2 = (const float*)d_in[base + 5];
    const float* g1  = (const float*)d_in[base + 6];
    const float* g2  = (const float*)d_in[base + 7];

    int H2 = h * 2;
    k_upsample2<<<EWGRID(N_IMG * cinF * H2 * H2)>>>(feats, ws + OFF_UP,
                                                    N_IMG * cinF, h, h);
    int f = 64 / H2;
    const float* lay;
    if (f > 1) {
      k_avgpool<<<EWGRID(N_IMG * GDIM * H2 * H2)>>>(ws + OFF_LAYOUT, ws + OFF_LAY,
                                                    N_IMG * GDIM, 64, 64, f);
      lay = ws + OFF_LAY;
    } else {
      lay = ws + OFF_LAYOUT;
    }
    int cin = GDIM + cinF, cout = couts[m];
    k_concat2<<<EWGRID(N_IMG * cin * H2 * H2)>>>(lay, ws + OFF_UP, ws + OFF_CC,
                                                 N_IMG, GDIM, cinF, H2 * H2);
    conv(ws + OFF_CC, W1, b1, ws + OFF_X, cin, H2, cout, 3, 1, 0);
    bnleaky(ws + OFF_X, g1, be1, cout, H2 * H2);
    conv(ws + OFF_X, W2, b2, feats, cout, H2, cout, 3, 1, 0);
    bnleaky(feats, g2, be2, cout, H2 * H2);
    cinF = cout; h = H2;
  }

  // ---- out_conv: leaky(3x3 conv) then 1x1 conv -> img ----
  conv(feats, ocW1, ocb1, ws + OFF_UP, 64, 64, 64, 3, 1, 2);
  conv(ws + OFF_UP, ocW2, ocb2, outp, 64, 64, 3, 1, 0, 0);
}
